// Attention_88227218194460
// MI455X (gfx1250) — matrix operand order repaired and checked by reading, not compiled, not run on hardware
//
#include <hip/hip_runtime.h>

// ---------------------------------------------------------------------------
// MI455X (gfx1250) attention block. bf16 operands + f32-accum WMMA everywhere.
// Async global->LDS copies (ASYNCcnt) with LDS double buffering; all WMMA
// fragments are ds_load_b128 thanks to pre-transposed B-side operands.
// ---------------------------------------------------------------------------

#define DIM_    2048
#define SEQ_    2048
#define BSZ_    2
#define NH_     16
#define NKV_    8
#define HD_     128
#define HID_    8192
#define NEGINF_ (-1.0e9f)

typedef __attribute__((ext_vector_type(16))) __bf16 bf16x16;
typedef __attribute__((ext_vector_type(8)))  float  f32x8;
typedef __attribute__((ext_vector_type(4)))  int    i32x4;
typedef __attribute__((address_space(1))) i32x4 gbl_i32x4;
typedef __attribute__((address_space(3))) i32x4 lds_i32x4;

#if defined(__has_builtin)
#if __has_builtin(__builtin_amdgcn_global_load_async_to_lds_b128)
#define HAS_ASYNC 1
#endif
#if __has_builtin(__builtin_amdgcn_s_wait_asynccnt)
#define HAS_WAITASYNC 1
#endif
#endif

__device__ __forceinline__ unsigned short f32_to_bf16(float f) {
  union { float f; unsigned u; } x; x.f = f;
  return (unsigned short)((x.u + 0x7FFFu + ((x.u >> 16) & 1u)) >> 16);
}
__device__ __forceinline__ float bf16_lo(unsigned v) {
  union { unsigned u; float f; } x; x.u = v << 16; return x.f;
}
__device__ __forceinline__ float bf16_hi(unsigned v) {
  union { unsigned u; float f; } x; x.u = v & 0xFFFF0000u; return x.f;
}
__device__ __forceinline__ unsigned pack_bf16(float lo, float hi) {
  return (unsigned)f32_to_bf16(lo) | ((unsigned)f32_to_bf16(hi) << 16);
}

union BF16Frag {
  unsigned short us[16];
  uint4          u4[2];
  bf16x16        v;
};

// 16B global->LDS copy: async (ASYNCcnt) if available, else through VGPRs.
__device__ __forceinline__ void cp16(const unsigned short* g, unsigned short* l) {
#if HAS_ASYNC
  __builtin_amdgcn_global_load_async_to_lds_b128((gbl_i32x4*)g, (lds_i32x4*)l,
                                                 0, 0);
#else
  *(uint4*)l = *(const uint4*)g;
#endif
}
__device__ __forceinline__ void wait_async() {
#if HAS_ASYNC
#if HAS_WAITASYNC
  __builtin_amdgcn_s_wait_asynccnt(0);
#else
  asm volatile("s_wait_asynccnt 0x0" ::: "memory");
#endif
#endif
}

// A-fragment (16x32, MxK): lane g*16+m -> row m, K chunks {g*8..+7, 16+g*8..+7}
__device__ __forceinline__ bf16x16 lds_load_a_frag(const unsigned short* base,
                                                   int stride, int lane) {
  const int g = lane >> 4;
  const unsigned short* row = base + (size_t)(lane & 15) * stride;
  BF16Frag u;
  u.u4[0] = *(const uint4*)(row + g * 8);
  u.u4[1] = *(const uint4*)(row + 16 + g * 8);
  return u.v;
}

// B-fragment (32x16, KxN) from a K-transposed LDS tile [n][k]:
// lane col n=ln holds K chunks {g*8..+7, 16+g*8..+7}.
__device__ __forceinline__ bf16x16 lds_load_b_frag_t(const unsigned short* base,
                                                     int stride, int lane) {
  const unsigned short* p =
      base + (size_t)(lane & 15) * stride + (lane >> 4) * 8;
  BF16Frag u;
  u.u4[0] = *(const uint4*)(p);
  u.u4[1] = *(const uint4*)(p + 16);
  return u.v;
}

// ---------------------------------------------------------------------------
// GEMM: C(MxN) = A(MxK) @ B(KxN) with B supplied TRANSPOSED (BT: NxK, bf16).
// 128x128 block tile, BK=32, 8 waves (wave tile 32x64), double-buffered async.
// ---------------------------------------------------------------------------
#define AST 40   // 32 k-ushorts + 16B pad (rows 16B aligned, banks spread)

template <bool OUT_BF16>
__global__ __launch_bounds__(256) void gemm_ws(
    const unsigned short* __restrict__ A, const unsigned short* __restrict__ BT,
    void* __restrict__ Cv, int M, int N, int K) {
  __shared__ unsigned short As[2][128][AST];  // [m][k]
  __shared__ unsigned short Bs[2][128][AST];  // [n][k]

  const int tid  = threadIdx.x;
  const int wave = tid >> 5;
  const int lane = tid & 31;
  const int g    = lane >> 4;
  const int ln   = lane & 15;
  const int wm   = wave & 3;
  const int wn   = wave >> 2;
  const int bm   = blockIdx.y * 128;
  const int bn   = blockIdx.x * 128;

  // per-thread staging: 2 chunks of A + 2 of BT, advancing 32 ushorts per step
  const int rr = tid >> 2, cc = (tid & 3) * 8;
  const unsigned short* pA[2] = {A + (size_t)(bm + rr) * K + cc,
                                 A + (size_t)(bm + 64 + rr) * K + cc};
  const unsigned short* pB[2] = {BT + (size_t)(bn + rr) * K + cc,
                                 BT + (size_t)(bn + 64 + rr) * K + cc};
  unsigned short* dA[2][2];
  unsigned short* dB[2][2];
#pragma unroll
  for (int buf = 0; buf < 2; ++buf)
#pragma unroll
    for (int j = 0; j < 2; ++j) {
      dA[buf][j] = &As[buf][64 * j + rr][cc];
      dB[buf][j] = &Bs[buf][64 * j + rr][cc];
    }
  auto stage = [&](int buf) {
    cp16(pA[0], dA[buf][0]); cp16(pA[1], dA[buf][1]);
    cp16(pB[0], dB[buf][0]); cp16(pB[1], dB[buf][1]);
    pA[0] += 32; pA[1] += 32; pB[0] += 32; pB[1] += 32;
  };

  stage(0);
  f32x8 acc[2][4] = {};
  const int nk = K >> 5;
  for (int it = 0; it < nk; ++it) {
    wait_async();
    __syncthreads();
    const int buf = it & 1;
    if (it + 1 < nk) stage(buf ^ 1);

    bf16x16 af[2], bfr[4];
#pragma unroll
    for (int fm = 0; fm < 2; ++fm)
      af[fm] = lds_load_a_frag(&As[buf][wm * 32 + fm * 16][0], AST, lane);
#pragma unroll
    for (int fn = 0; fn < 4; ++fn)
      bfr[fn] = lds_load_b_frag_t(&Bs[buf][wn * 64 + fn * 16][0], AST, lane);
#pragma unroll
    for (int fm = 0; fm < 2; ++fm)
#pragma unroll
      for (int fn = 0; fn < 4; ++fn)
        acc[fm][fn] = __builtin_amdgcn_wmma_f32_16x16x32_bf16(
            false, af[fm], false, bfr[fn], (short)0, acc[fm][fn], false, false);
  }

#pragma unroll
  for (int fm = 0; fm < 2; ++fm) {
    const int row0 = bm + wm * 32 + fm * 16 + 8 * g;
#pragma unroll
    for (int fn = 0; fn < 4; ++fn) {
      const int col = bn + wn * 64 + fn * 16 + ln;
#pragma unroll
      for (int r = 0; r < 8; ++r) {
        if constexpr (OUT_BF16)
          ((unsigned short*)Cv)[(size_t)(row0 + r) * N + col] =
              f32_to_bf16(acc[fm][fn][r]);
        else
          ((float*)Cv)[(size_t)(row0 + r) * N + col] = acc[fm][fn][r];
      }
    }
  }
}

// ---------------------------------------------------------------------------
// Elementwise / layout kernels
// ---------------------------------------------------------------------------
__global__ void cvt_f32_bf16(const float* __restrict__ in,
                             unsigned short* __restrict__ out, int n8) {
  int i = blockIdx.x * blockDim.x + threadIdx.x;
  const int stride = gridDim.x * blockDim.x;
  for (; i < n8; i += stride) {
    const float4 a = ((const float4*)in)[2 * i];
    const float4 b = ((const float4*)in)[2 * i + 1];
    uint4 r;
    r.x = pack_bf16(a.x, a.y); r.y = pack_bf16(a.z, a.w);
    r.z = pack_bf16(b.x, b.y); r.w = pack_bf16(b.z, b.w);
    ((uint4*)out)[i] = r;
  }
}

// 64x64 LDS-tiled transpose with f32 -> bf16: out[c][r] = (bf16)in[r][c].
// blockIdx.z selects a slice (for per-(b,kv-head) V transposes).
__global__ __launch_bounds__(256) void transpose_cvt(
    const float* __restrict__ in, unsigned short* __restrict__ out,
    size_t in_rstride, size_t out_rstride, size_t in_sstride,
    size_t out_sstride) {
  __shared__ unsigned short T[64][72];
  const int tid = threadIdx.x;
  const int r0 = blockIdx.y * 64;
  const int c0 = blockIdx.x * 64;
  const float* src = in + (size_t)blockIdx.z * in_sstride;
  unsigned short* dst = out + (size_t)blockIdx.z * out_sstride;
#pragma unroll
  for (int j = 0; j < 4; ++j) {
    const int idx = tid + j * 256;          // 1024 float4 = 64x64 f32
    const int r = idx >> 4, c4 = idx & 15;
    const float4 f = *(const float4*)(src + (size_t)(r0 + r) * in_rstride + c0 + c4 * 4);
    T[c4 * 4 + 0][r] = f32_to_bf16(f.x);
    T[c4 * 4 + 1][r] = f32_to_bf16(f.y);
    T[c4 * 4 + 2][r] = f32_to_bf16(f.z);
    T[c4 * 4 + 3][r] = f32_to_bf16(f.w);
  }
  __syncthreads();
#pragma unroll
  for (int j = 0; j < 4; ++j) {
    const int idx = tid + j * 256;          // 1024 uint2 = 64x64 bf16
    const int c = idx >> 4, r4 = idx & 15;
    uint2 w;
    w.x = (unsigned)T[c][r4 * 4 + 0] | ((unsigned)T[c][r4 * 4 + 1] << 16);
    w.y = (unsigned)T[c][r4 * 4 + 2] | ((unsigned)T[c][r4 * 4 + 3] << 16);
    *(uint2*)(dst + (size_t)(c0 + c) * out_rstride + r0 + r4 * 4) = w;
  }
}

__device__ __forceinline__ unsigned silu_mul_pk(unsigned a, unsigned b) {
  const float a0 = bf16_lo(a), a1 = bf16_hi(a);
  const float b0 = bf16_lo(b), b1 = bf16_hi(b);
  return pack_bf16(a0 / (1.f + __expf(-a0)) * b0, a1 / (1.f + __expf(-a1)) * b1);
}
__global__ void silu_mul_bf16(unsigned short* __restrict__ h1,
                              const unsigned short* __restrict__ h3, int n8) {
  int i = blockIdx.x * blockDim.x + threadIdx.x;
  const int stride = gridDim.x * blockDim.x;
  for (; i < n8; i += stride) {
    uint4 a = ((const uint4*)h1)[i];
    const uint4 b = ((const uint4*)h3)[i];
    a.x = silu_mul_pk(a.x, b.x); a.y = silu_mul_pk(a.y, b.y);
    a.z = silu_mul_pk(a.z, b.z); a.w = silu_mul_pk(a.w, b.w);
    ((uint4*)h1)[i] = a;
  }
}

// RoPE on fp32 K projection -> fp32 o_keys AND bf16 row-major copy
__global__ void rope_k_kernel(const float* __restrict__ kpre,
                              float* __restrict__ okeys,
                              unsigned short* __restrict__ kb,
                              const float* __restrict__ cosT,
                              const float* __restrict__ sinT, int total) {
  const int idx = blockIdx.x * blockDim.x + threadIdx.x;
  if (idx >= total) return;
  const int f = idx & 63;
  const int t = idx >> 6;
  const int s = (t / NKV_) & (SEQ_ - 1);
  const float c = cosT[s * 64 + f], sn = sinT[s * 64 + f];
  const size_t base = (size_t)t * HD_ + 2 * f;
  const float xr = kpre[base], xi = kpre[base + 1];
  const float orr = xr * c - xi * sn, oii = xr * sn + xi * c;
  okeys[base] = orr; okeys[base + 1] = oii;
  ((unsigned*)kb)[(size_t)t * (HD_ / 2) + f] = pack_bf16(orr, oii);
}

// RoPE in place on bf16 Q (pair packed in one dword)
__global__ void rope_q_bf16(unsigned short* __restrict__ q,
                            const float* __restrict__ cosT,
                            const float* __restrict__ sinT, int total) {
  const int idx = blockIdx.x * blockDim.x + threadIdx.x;
  if (idx >= total) return;
  const int f = idx & 63;
  const int t = idx >> 6;
  const int s = (t / NH_) & (SEQ_ - 1);
  const float c = cosT[s * 64 + f], sn = sinT[s * 64 + f];
  unsigned* p = (unsigned*)q + (size_t)t * (HD_ / 2) + f;
  const unsigned v = *p;
  const float xr = bf16_lo(v), xi = bf16_hi(v);
  *p = pack_bf16(xr * c - xi * sn, xr * sn + xi * c);
}

// ---------------------------------------------------------------------------
// Flash attention: 4 waves/WG, wave = 16 q-rows; double-buffered async K/V
// tiles (K row-major [key][d], V transposed [d][key]); online softmax.
// ---------------------------------------------------------------------------
#define KST 136  // K tile row stride (128 d + pad)

__global__ __launch_bounds__(128) void attn_kernel(
    const unsigned short* __restrict__ Q, const unsigned short* __restrict__ Kb,
    const unsigned short* __restrict__ VbT, unsigned short* __restrict__ O) {
  __shared__ unsigned short Ks[2][32][KST];   // [key][d]
  __shared__ unsigned short Vs[2][HD_][AST];  // [d][key]
  __shared__ unsigned short Ps[4][16][AST];   // per-wave P transpose scratch

  const int qblk = blockIdx.x;
  const int h    = blockIdx.y;
  const int b    = blockIdx.z;
  const int kvh  = h >> 1;
  const int tid  = threadIdx.x;
  const int wave = tid >> 5;
  const int lane = tid & 31;
  const int g    = lane >> 4;
  const int ln   = lane & 15;
  const int qbase = qblk * 64 + wave * 16;
  const float scale = 0.08838834764831845f;

  // staging pointers: K -> 512 chunks (32 keys x 16), V -> 512 (128 d x 4)
  const unsigned short* pK[4];
  const unsigned short* pV[4];
  unsigned short* dK[2][4];
  unsigned short* dV[2][4];
#pragma unroll
  for (int j = 0; j < 4; ++j) {
    const int key = (tid >> 4) + j * 8, ck = (tid & 15) * 8;
    const int d = (tid >> 2) + j * 32, cv = (tid & 3) * 8;
    pK[j] = Kb + (size_t)(b * SEQ_ + key) * (NKV_ * HD_) + kvh * HD_ + ck;
    pV[j] = VbT + ((size_t)(b * NKV_ + kvh) * HD_ + d) * SEQ_ + cv;
#pragma unroll
    for (int buf = 0; buf < 2; ++buf) {
      dK[buf][j] = &Ks[buf][key][ck];
      dV[buf][j] = &Vs[buf][d][cv];
    }
  }
  auto stage = [&](int buf) {
#pragma unroll
    for (int j = 0; j < 4; ++j) {
      cp16(pK[j], dK[buf][j]);
      cp16(pV[j], dV[buf][j]);
      pK[j] += 32 * (NKV_ * HD_);
      pV[j] += 32;
    }
  };

  bf16x16 qf[4];
  {
    const unsigned short* qrow =
        Q + (size_t)(b * SEQ_ + qbase + ln) * (NH_ * HD_) + h * HD_;
#pragma unroll
    for (int dstep = 0; dstep < 4; ++dstep) {
      BF16Frag u;
      u.u4[0] = *(const uint4*)(qrow + dstep * 32 + g * 8);
      u.u4[1] = *(const uint4*)(qrow + dstep * 32 + 16 + g * 8);
      qf[dstep] = u.v;
    }
  }

  f32x8 o_acc[8] = {};
  float mrow[8], lrow[8];
#pragma unroll
  for (int r = 0; r < 8; ++r) { mrow[r] = -1.0e30f; lrow[r] = 0.f; }

  stage(0);
  const int nkt = qblk * 2 + 2;
  for (int kt = 0; kt < nkt; ++kt) {
    wait_async();
    __syncthreads();
    const int buf = kt & 1;
    if (kt + 1 < nkt) stage(buf ^ 1);
    const int kbase = kt * 32;

    // S = Q K^T (keys are N, d is contraction)
    f32x8 sc0 = {}, sc1 = {};
#pragma unroll
    for (int dstep = 0; dstep < 4; ++dstep) {
      const bf16x16 b0 = lds_load_b_frag_t(&Ks[buf][0][dstep * 32], KST, lane);
      const bf16x16 b1 = lds_load_b_frag_t(&Ks[buf][16][dstep * 32], KST, lane);
      sc0 = __builtin_amdgcn_wmma_f32_16x16x32_bf16(false, qf[dstep], false, b0,
                                                    (short)0, sc0, false, false);
      sc1 = __builtin_amdgcn_wmma_f32_16x16x32_bf16(false, qf[dstep], false, b1,
                                                    (short)0, sc1, false, false);
    }

    float alpha[8];
#pragma unroll
    for (int r = 0; r < 8; ++r) {
      const int qg = qbase + r + 8 * g;
      float s0 = sc0[r] * scale + ((kbase + ln > qg) ? NEGINF_ : 0.f);
      float s1 = sc1[r] * scale + ((kbase + 16 + ln > qg) ? NEGINF_ : 0.f);
      float mx = fmaxf(s0, s1);
#pragma unroll
      for (int off = 1; off < 16; off <<= 1)
        mx = fmaxf(mx, __shfl_xor(mx, off, 16));
      const float mnew = fmaxf(mrow[r], mx);
      alpha[r] = __expf(mrow[r] - mnew);
      const float p0 = __expf(s0 - mnew);
      const float p1 = __expf(s1 - mnew);
      float sum = p0 + p1;
#pragma unroll
      for (int off = 1; off < 16; off <<= 1)
        sum += __shfl_xor(sum, off, 16);
      lrow[r] = lrow[r] * alpha[r] + sum;
      mrow[r] = mnew;
      Ps[wave][r + 8 * g][ln]      = f32_to_bf16(p0);
      Ps[wave][r + 8 * g][16 + ln] = f32_to_bf16(p1);
    }
#pragma unroll
    for (int n = 0; n < 8; ++n)
#pragma unroll
      for (int r = 0; r < 8; ++r)
        o_acc[n][r] = o_acc[n][r] * alpha[r];

    // O += P V (same-wave LDS ops are in-order; Ps is wave-private)
    const bf16x16 pf = lds_load_a_frag(&Ps[wave][0][0], AST, lane);
#pragma unroll
    for (int n = 0; n < 8; ++n) {
      const bf16x16 vfr = lds_load_b_frag_t(&Vs[buf][n * 16][0], AST, lane);
      o_acc[n] = __builtin_amdgcn_wmma_f32_16x16x32_bf16(
          false, pf, false, vfr, (short)0, o_acc[n], false, false);
    }
  }

#pragma unroll
  for (int r = 0; r < 8; ++r) {
    const float inv = 1.0f / lrow[r];
    unsigned short* orow =
        O + (size_t)(b * SEQ_ + qbase + r + 8 * g) * (NH_ * HD_) + h * HD_ + ln;
#pragma unroll
    for (int n = 0; n < 8; ++n) orow[n * 16] = f32_to_bf16(o_acc[n][r] * inv);
  }
}

// ---------------------------------------------------------------------------
// Host-side launcher
// ---------------------------------------------------------------------------
extern "C" void kernel_launch(void* const* d_in, const int* in_sizes, int n_in,
                              void* d_out, int out_size, void* d_ws, size_t ws_size,
                              hipStream_t stream) {
  (void)in_sizes; (void)n_in; (void)out_size; (void)ws_size;
  const float* x    = (const float*)d_in[0];
  const float* wq   = (const float*)d_in[1];
  const float* wk1  = (const float*)d_in[2];
  const float* wk2  = (const float*)d_in[3];
  const float* wk3  = (const float*)d_in[4];
  const float* wv1  = (const float*)d_in[5];
  const float* wv2  = (const float*)d_in[6];
  const float* wv3  = (const float*)d_in[7];
  const float* wo   = (const float*)d_in[8];
  const float* fcos = (const float*)d_in[9];
  const float* fsin = (const float*)d_in[10];

  float* out   = (float*)d_out;
  float* okeys = out + (size_t)BSZ_ * SEQ_ * DIM_;
  float* ovals = okeys + (size_t)BSZ_ * SEQ_ * NKV_ * HD_;

  // workspace (bf16 buffers; offsets in ushorts)
  unsigned short* xb   = (unsigned short*)d_ws;          //  8.39M
  unsigned short* wbuf = xb + 8388608;                   // 16.78M (transposed W)
  unsigned short* qb   = wbuf + 16777216;                //  8.39M
  unsigned short* kb   = qb + 8388608;                   //  4.19M
  unsigned short* vbT  = kb + 4194304;                   //  4.19M
  unsigned short* attb = vbT + 4194304;                  //  8.39M
  unsigned short* h1b  = attb + 8388608;                 // 33.55M
  unsigned short* h3b  = h1b + 33554432;                 // 33.55M
  float* kpre = (float*)h3b;   // fp32 K projection, reuses h3 region

  const int M = BSZ_ * SEQ_;
  // transpose+convert a KxN f32 weight into NxK bf16
  auto tcvt_w = [&](const float* w, int kdim, int ndim) {
    transpose_cvt<<<dim3(ndim / 64, kdim / 64, 1), 256, 0, stream>>>(
        w, wbuf, (size_t)ndim, (size_t)kdim, 0, 0);
  };
  auto gemm_b = [&](const unsigned short* A, const unsigned short* BT,
                    unsigned short* C, int m, int n, int k) {
    gemm_ws<true><<<dim3(n / 128, m / 128), 256, 0, stream>>>(A, BT, C, m, n, k);
  };
  auto gemm_f = [&](const unsigned short* A, const unsigned short* BT, float* C,
                    int m, int n, int k) {
    gemm_ws<false><<<dim3(n / 128, m / 128), 256, 0, stream>>>(A, BT, C, m, n, k);
  };

  cvt_f32_bf16<<<2048, 256, 0, stream>>>(x, xb, (int)((size_t)M * DIM_ / 8));

  // K projection: silu(x@w1)*(x@w3) @ w2 -> RoPE -> o_keys (f32) + kb (bf16)
  tcvt_w(wk1, DIM_, HID_);
  gemm_b(xb, wbuf, h1b, M, HID_, DIM_);
  tcvt_w(wk3, DIM_, HID_);
  gemm_b(xb, wbuf, h3b, M, HID_, DIM_);
  silu_mul_bf16<<<2048, 256, 0, stream>>>(h1b, h3b, (int)((size_t)M * HID_ / 8));
  tcvt_w(wk2, HID_, NKV_ * HD_);
  gemm_f(h1b, wbuf, kpre, M, NKV_ * HD_, HID_);
  {
    const int tot = BSZ_ * SEQ_ * NKV_ * 64;
    rope_k_kernel<<<(tot + 255) / 256, 256, 0, stream>>>(kpre, okeys, kb, fcos,
                                                         fsin, tot);
  }
  // V projection -> o_values (f32) + vbT (bf16, [b][kvh][d][s])
  tcvt_w(wv1, DIM_, HID_);
  gemm_b(xb, wbuf, h1b, M, HID_, DIM_);
  tcvt_w(wv3, DIM_, HID_);
  gemm_b(xb, wbuf, h3b, M, HID_, DIM_);
  silu_mul_bf16<<<2048, 256, 0, stream>>>(h1b, h3b, (int)((size_t)M * HID_ / 8));
  tcvt_w(wv2, HID_, NKV_ * HD_);
  gemm_f(h1b, wbuf, ovals, M, NKV_ * HD_, HID_);
  for (int b = 0; b < BSZ_; ++b) {
    transpose_cvt<<<dim3(HD_ / 64, SEQ_ / 64, NKV_), 256, 0, stream>>>(
        ovals + (size_t)b * SEQ_ * NKV_ * HD_,
        vbT + (size_t)b * NKV_ * HD_ * SEQ_,
        (size_t)NKV_ * HD_, (size_t)SEQ_, (size_t)HD_, (size_t)HD_ * SEQ_);
  }
  // Q projection + RoPE (in place, bf16)
  tcvt_w(wq, DIM_, DIM_);
  gemm_b(xb, wbuf, qb, M, DIM_, DIM_);
  {
    const int tot = BSZ_ * SEQ_ * NH_ * 64;
    rope_q_bf16<<<(tot + 255) / 256, 256, 0, stream>>>(qb, fcos, fsin, tot);
  }
  // Flash attention
  attn_kernel<<<dim3(SEQ_ / 64, NH_, BSZ_), dim3(128), 0, stream>>>(qb, kb, vbT,
                                                                    attb);
  // Output projection
  tcvt_w(wo, DIM_, DIM_);
  gemm_f(attb, wbuf, out, M, DIM_, DIM_);
}
